// MTP_11527692222730
// MI455X (gfx1250) — matrix operand order repaired and checked by reading, not compiled, not run on hardware
//
#include <hip/hip_runtime.h>
#include <hip/hip_bf16.h>
#include <stdint.h>

// ---------------- problem constants (match reference) ----------------
#define NHEADS 2
#define DH 1024
#define DV 50257
#define BB 2
#define TT 2048
#define NROWS (BB * TT)                       // 4096
#define LBL_STRIDE (TT + NHEADS)              // 2050
#define STATS_CHUNK 1024
#define NSCHUNK ((DV + STATS_CHUNK - 1) / STATS_CHUNK)   // 50
#define GCHUNK 128
#define NGCHUNK ((DV + GCHUNK - 1) / GCHUNK)             // 393
#define VPAD (NGCHUNK * GCHUNK)                          // 50304 (zero-padded V)
#define INV_ROWS (1.0f / (float)NROWS)
#define NEG_BIG (-3.0e38f)

// grad kernel LDS layout (dynamic shared only -> base offset 0)
#define XL_OFF 0                               // 16 x 1024 bf16 = 32768 B
#define PL_OFF (16 * DH * 2)                   // 16 x 128 bf16  -> 32768
#define GRAD_LDS (PL_OFF + 16 * GCHUNK * 2)    // 36864 B
// stats kernel LDS layout (dynamic shared only)
#define SXL_OFF 0                              // 16 x 1024 bf16 = 32768 B
#define SMX_OFF (16 * DH * 2)
#define SMS_OFF (SMX_OFF + 8 * 16 * 4)
#define STATS_LDS (SMS_OFF + 8 * 16 * 4)       // 33792 B

// ---------------- WMMA / TDM types ----------------
typedef __attribute__((ext_vector_type(16))) __bf16 v16bf;
typedef __attribute__((ext_vector_type(8)))  float  v8f;
typedef __attribute__((ext_vector_type(4)))  unsigned int v4u;
typedef __attribute__((ext_vector_type(8)))  int v8i;
typedef __attribute__((ext_vector_type(4)))  int v4i;

union Frag16 {
    v16bf v;
    uint4 q[2];
    unsigned short h[16];
};

static __device__ __forceinline__ unsigned short f2bf(float f) {
    union { float f; uint32_t u; } c; c.f = f;
    uint32_t u = c.u;
    uint32_t r = u + 0x7FFFu + ((u >> 16) & 1u);   // round-to-nearest-even
    if ((u & 0x7F800000u) == 0x7F800000u) r = u;   // inf/nan passthrough
    return (unsigned short)(r >> 16);
}

static __device__ __forceinline__ float halfwave_max(float x) {
    for (int off = 1; off < 16; off <<= 1) x = fmaxf(x, __shfl_xor(x, off, 32));
    return x;
}
static __device__ __forceinline__ float halfwave_sum(float x) {
    for (int off = 1; off < 16; off <<= 1) x += __shfl_xor(x, off, 32);
    return x;
}

static __device__ __forceinline__ v8f wmma_bf16(const v16bf& a, const v16bf& b, const v8f& c) {
    return __builtin_amdgcn_wmma_f32_16x16x32_bf16(false, a, false, b, (short)0, c, false, false);
}

// ---------------- Tensor Data Mover helpers (guarded) ----------------
#if defined(__gfx1250__) && __has_builtin(__builtin_amdgcn_tensor_load_to_lds)
#define HAVE_TDM 1
#else
#define HAVE_TDM 0
#endif

#if HAVE_TDM
static __device__ __forceinline__ void tdm_wait0() {
#if __has_builtin(__builtin_amdgcn_s_wait_tensorcnt)
    __builtin_amdgcn_s_wait_tensorcnt(0);
#else
    asm volatile("s_wait_tensorcnt 0x0" ::: "memory");
#endif
}

// 2D tile load: rows of (row_u8 * 8) bytes, tile_rows rows, tensor_rows valid
// rows remaining in the tensor (rows past that are hardware-zero-filled).
static __device__ __forceinline__ void tdm_load_2d(unsigned lds_off, const void* g,
                                                   unsigned tensor_rows, unsigned row_u8,
                                                   unsigned tile_rows) {
    uint64_t ga = (uint64_t)(uintptr_t)g;
    v4u g0;
    g0[0] = 1u;                                             // count=1 (user D#)
    g0[1] = lds_off;                                        // lds_addr (bytes)
    g0[2] = (unsigned)ga;                                   // global_addr[31:0]
    g0[3] = (unsigned)((ga >> 32) & 0x01FFFFFFu) | (2u << 30); // addr[56:32] | type=2
    v8i g1;
    g1[0] = (int)(3u << 16);                                // data_size = 8B units
    g1[1] = (int)((row_u8 & 0xFFFFu) << 16);                // tensor_dim0[15:0]
    g1[2] = (int)(((row_u8 >> 16) & 0xFFFFu) |              // tensor_dim0[31:16]
                  ((tensor_rows & 0xFFFFu) << 16));         // tensor_dim1[15:0]
    g1[3] = (int)(((tensor_rows >> 16) & 0xFFFFu) |         // tensor_dim1[31:16]
                  ((row_u8 & 0xFFFFu) << 16));              // tile_dim0
    g1[4] = (int)(tile_rows & 0xFFFFu);                     // tile_dim1 (tile_dim2=0)
    g1[5] = (int)row_u8;                                    // tensor_dim0_stride[31:0]
    g1[6] = 0;
    g1[7] = 0;
    v4i gz = {0, 0, 0, 0};
#if __has_include(<hip/amd_detail/amd_gfx1250_TDM.h>)
    v8i gz8 = {0, 0, 0, 0, 0, 0, 0, 0};
    __builtin_amdgcn_tensor_load_to_lds(g0, g1, gz, gz, gz8, 0);   // clang-23 6-arg form
#else
    __builtin_amdgcn_tensor_load_to_lds(g0, g1, gz, gz, 0);        // ROCm 7.2 5-arg form
#endif
}
#endif  // HAVE_TDM

// ---------------- prep kernels ----------------
__global__ void cvt_bf16_kernel(const float* __restrict__ src,
                                unsigned short* __restrict__ dst, int n) {
    int i = blockIdx.x * blockDim.x + threadIdx.x;
    if (i < n) dst[i] = f2bf(src[i]);
}

__global__ void zero_f32_kernel(float* __restrict__ p, int n) {
    int i = blockIdx.x * blockDim.x + threadIdx.x;
    if (i < n) p[i] = 0.0f;
}

// Wbt[head][h][v] = bf16(W[head][v][h]); v padded to VPAD with zeros.
// grid: (VPAD/32, DH/32, NHEADS), block: (32, 8)
__global__ void transpose_w_kernel(const float* __restrict__ W,
                                   unsigned short* __restrict__ Wbt) {
    __shared__ float tile[32][33];
    const int v0 = blockIdx.x * 32, h0 = blockIdx.y * 32, head = blockIdx.z;
    const int tx = threadIdx.x, ty = threadIdx.y;
#pragma unroll
    for (int j = 0; j < 4; ++j) {
        const int v = v0 + ty + j * 8;
        tile[ty + j * 8][tx] = (v < DV) ? W[((size_t)head * DV + v) * DH + h0 + tx] : 0.0f;
    }
    __syncthreads();
#pragma unroll
    for (int j = 0; j < 4; ++j) {
        const int h = h0 + ty + j * 8;
        Wbt[((size_t)head * DH + h) * VPAD + v0 + tx] = f2bf(tile[tx][ty + j * 8]);
    }
}

// ---------------- pass 1: per-(row, chunk) softmax statistics ----------------
// grid: (NROWS/16, NSCHUNK, NHEADS), block: 256 (8 waves), dyn LDS = STATS_LDS
__global__ void stats_kernel(const unsigned short* __restrict__ Xb,
                             const unsigned short* __restrict__ Wb,
                             const float* __restrict__ biases,
                             float* __restrict__ pm, float* __restrict__ ps) {
    extern __shared__ __align__(16) char smem_raw[];
    unsigned short* Xl = (unsigned short*)(smem_raw + SXL_OFF);
    float* smx = (float*)(smem_raw + SMX_OFF);
    float* sms = (float*)(smem_raw + SMS_OFF);

    const int rb = blockIdx.x, chunk = blockIdx.y, head = blockIdx.z;
    const int tid = threadIdx.x, wave = tid >> 5, lane = tid & 31;
    const int n16 = lane & 15;
    const bool lo = (lane < 16);
    const int hi8 = lo ? 0 : 8;

#if HAVE_TDM
    if (wave == 0) {
        tdm_load_2d(SXL_OFF, Xb + (size_t)rb * 16 * DH, 16u, DH * 2 / 8, 16u);
        tdm_wait0();
    }
#else
    {
        const uint4* src = (const uint4*)(Xb + (size_t)rb * 16 * DH);
        uint4* dst = (uint4*)Xl;
        for (int i = tid; i < 16 * DH / 8; i += 256) dst[i] = src[i];
    }
#endif
    __syncthreads();

    const unsigned short* Wh = Wb + (size_t)head * DV * DH;
    const int ab1 = lo ? 0 : 16, ab2 = lo ? 8 : 24, kbo = lo ? 0 : 16;

    float runM[8], runS[8];
#pragma unroll
    for (int r = 0; r < 8; ++r) { runM[r] = NEG_BIG; runS[r] = 0.0f; }

#pragma unroll 1
    for (int it = 0; it < 8; ++it) {
        const int v = chunk * STATS_CHUNK + it * 128 + wave * 16 + n16;
        const int vc = (v < DV) ? v : (DV - 1);
        const bool valid = (v < DV);

        // two independent accumulator chains for WMMA/load overlap
        v8f c0 = {0.f, 0.f, 0.f, 0.f, 0.f, 0.f, 0.f, 0.f};
        v8f c1 = c0;
        const unsigned short* wrow = Wh + (size_t)vc * DH;
#pragma unroll 2
        for (int k0 = 0; k0 < DH; k0 += 64) {
            Frag16 a0, b0, a1, b1;
            const unsigned short* xr = Xl + n16 * DH + k0;
            a0.q[0] = *(const uint4*)(xr + ab1);
            a0.q[1] = *(const uint4*)(xr + ab2);
            a1.q[0] = *(const uint4*)(xr + 32 + ab1);
            a1.q[1] = *(const uint4*)(xr + 32 + ab2);
            const unsigned short* wp = wrow + k0 + kbo;
            b0.q[0] = *(const uint4*)(wp);
            b0.q[1] = *(const uint4*)(wp + 8);
            b1.q[0] = *(const uint4*)(wp + 32);
            b1.q[1] = *(const uint4*)(wp + 40);
            c0 = wmma_bf16(a0.v, b0.v, c0);
            c1 = wmma_bf16(a1.v, b1.v, c1);
        }
        v8f c = c0 + c1;

        const float bias = valid ? biases[(size_t)head * DV + v] : 0.0f;
#pragma unroll
        for (int r = 0; r < 8; ++r) {
            float l = valid ? (c[r] + bias) : NEG_BIG;
            float m = halfwave_max(l);
            float e = valid ? __expf(l - m) : 0.0f;
            float s = halfwave_sum(e);
            float nm = fmaxf(runM[r], m);
            runS[r] = runS[r] * __expf(runM[r] - nm) + s * __expf(m - nm);
            runM[r] = nm;
        }
    }

    if (n16 == 0) {
#pragma unroll
        for (int r = 0; r < 8; ++r) {
            smx[wave * 16 + r + hi8] = runM[r];
            sms[wave * 16 + r + hi8] = runS[r];
        }
    }
    __syncthreads();

    if (tid < 16) {
        float M = NEG_BIG, S = 0.0f;
#pragma unroll
        for (int w = 0; w < 8; ++w) {
            float m2 = smx[w * 16 + tid], s2 = sms[w * 16 + tid];
            float nm = fmaxf(M, m2);
            S = S * __expf(M - nm) + s2 * __expf(m2 - nm);
            M = nm;
        }
        const size_t idx = ((size_t)head * NSCHUNK + chunk) * NROWS + rb * 16 + tid;
        pm[idx] = M;
        ps[idx] = S;
    }
}

// ---------------- pass 2: combine chunk stats -> logsumexp ----------------
__global__ void lse_kernel(const float* __restrict__ pm, const float* __restrict__ ps,
                           float* __restrict__ lse) {
    int i = blockIdx.x * blockDim.x + threadIdx.x;
    if (i >= NHEADS * NROWS) return;
    const int head = i >> 12, row = i & (NROWS - 1);
    float M = NEG_BIG, S = 0.0f;
    for (int c = 0; c < NSCHUNK; ++c) {
        const size_t idx = ((size_t)head * NSCHUNK + c) * NROWS + row;
        float m2 = pm[idx], s2 = ps[idx];
        float nm = fmaxf(M, m2);
        S = S * __expf(M - nm) + s2 * __expf(m2 - nm);
        M = nm;
    }
    lse[(size_t)head * NROWS + row] = M + __logf(S);
}

// ---------------- pass 3: label logit -> loss, and -onehot term of grad ----
// grid: (NROWS, NHEADS), block: 256
__global__ void label_kernel(const float* __restrict__ X, const int* __restrict__ labels,
                             const float* __restrict__ W, const float* __restrict__ biases,
                             const float* __restrict__ lse, float* __restrict__ out) {
    const int row = blockIdx.x, head = blockIdx.y;
    const int tid = threadIdx.x, wave = tid >> 5, lane = tid & 31;
    const int b = row >> 11, t = row & (TT - 1);
    int lbl = labels[b * LBL_STRIDE + t + 1 + head];
    if ((unsigned)lbl >= (unsigned)DV) lbl = 0;

    const float* wrow = W + ((size_t)head * DV + lbl) * DH;
    const float* xrow = X + (size_t)row * DH;
    float* grad = out + 1;

    float partial = 0.0f;
    for (int i = tid; i < DH; i += 256) {
        float wv = wrow[i];
        partial += xrow[i] * wv;
        atomicAdd(&grad[(size_t)row * DH + i], -wv * INV_ROWS);  // -onehot @ W
    }
    for (int off = 1; off < 32; off <<= 1) partial += __shfl_xor(partial, off, 32);
    __shared__ float red[8];
    if (lane == 0) red[wave] = partial;
    __syncthreads();
    if (tid == 0) {
        float ll = 0.0f;
        for (int w = 0; w < 8; ++w) ll += red[w];
        ll += biases[(size_t)head * DV + lbl];
        float contrib = (lse[(size_t)head * NROWS + row] - ll) * INV_ROWS;
        atomicAdd(&out[0], contrib);
    }
}

// ---------------- pass 4: dX += softmax/NROWS @ W (double-WMMA chain) ------
// Stage 1 streams W (v-major) for logits; stage 2 streams Wbt (h-major) so
// both WMMA operand fetches are contiguous 16B loads. No W tile in LDS.
// grid: (NROWS/16, NHEADS), block: 256 (8 waves), dynamic LDS = GRAD_LDS
__global__ void grad_kernel(const unsigned short* __restrict__ Xb,
                            const unsigned short* __restrict__ Wb,
                            const unsigned short* __restrict__ Wbt,
                            const float* __restrict__ biases,
                            const float* __restrict__ lse,
                            float* __restrict__ out) {
    extern __shared__ __align__(16) char smem_raw2[];
    unsigned short* Xl = (unsigned short*)(smem_raw2 + XL_OFF);   // 16 x 1024 bf16
    unsigned short* Pl = (unsigned short*)(smem_raw2 + PL_OFF);   // 16 x 128  bf16

    const int rb = blockIdx.x, head = blockIdx.y;
    const int tid = threadIdx.x, wave = tid >> 5, lane = tid & 31;
    const int n16 = lane & 15;
    const bool lo = (lane < 16);
    const int hi8 = lo ? 0 : 8;
    const int ab1 = lo ? 0 : 16, ab2 = lo ? 8 : 24, kbo = lo ? 0 : 16;

#if HAVE_TDM
    if (wave == 0) {
        tdm_load_2d(XL_OFF, Xb + (size_t)rb * 16 * DH, 16u, DH * 2 / 8, 16u);
        tdm_wait0();
    }
#else
    {
        const uint4* src = (const uint4*)(Xb + (size_t)rb * 16 * DH);
        uint4* dst = (uint4*)Xl;
        for (int i = tid; i < 16 * DH / 8; i += 256) dst[i] = src[i];
    }
#endif

    float lseR[8];
#pragma unroll
    for (int r = 0; r < 8; ++r)
        lseR[r] = lse[(size_t)head * NROWS + rb * 16 + hi8 + r];

    const unsigned short* Wh = Wb + (size_t)head * DV * DH;
    const unsigned short* Wth = Wbt + (size_t)head * DH * VPAD;
    const int hbase_w = wave * 128;

    v8f acc[8];
#pragma unroll
    for (int nt = 0; nt < 8; ++nt) acc[nt] = (v8f){0.f, 0.f, 0.f, 0.f, 0.f, 0.f, 0.f, 0.f};

#pragma unroll 1
    for (int ch = 0; ch < NGCHUNK; ++ch) {
        const int vbase = ch * GCHUNK;
        __syncthreads();   // previous iteration done reading Pl

        // ---- stage 1: recompute 16x16 logits tile, convert to P, park in LDS
        const int v = vbase + wave * 16 + n16;
        const int vc = (v < DV) ? v : (DV - 1);
        const bool valid = (v < DV);
        v8f c0 = {0.f, 0.f, 0.f, 0.f, 0.f, 0.f, 0.f, 0.f};
        v8f c1 = c0;
        const unsigned short* wrow = Wh + (size_t)vc * DH;
#pragma unroll 2
        for (int k0 = 0; k0 < DH; k0 += 64) {
            Frag16 a0, b0, a1, b1;
            const unsigned short* xr = Xl + n16 * DH + k0;
            a0.q[0] = *(const uint4*)(xr + ab1);
            a0.q[1] = *(const uint4*)(xr + ab2);
            a1.q[0] = *(const uint4*)(xr + 32 + ab1);
            a1.q[1] = *(const uint4*)(xr + 32 + ab2);
            const unsigned short* wp = wrow + k0 + kbo;
            b0.q[0] = *(const uint4*)(wp);
            b0.q[1] = *(const uint4*)(wp + 8);
            b1.q[0] = *(const uint4*)(wp + 32);
            b1.q[1] = *(const uint4*)(wp + 40);
            c0 = wmma_bf16(a0.v, b0.v, c0);
            c1 = wmma_bf16(a1.v, b1.v, c1);
        }
        v8f c = c0 + c1;
        const float bias = valid ? biases[(size_t)head * DV + v] : 0.0f;
#pragma unroll
        for (int r = 0; r < 8; ++r) {
            float p = valid ? __expf(c[r] + bias - lseR[r]) * INV_ROWS : 0.0f;
            Pl[(r + hi8) * GCHUNK + wave * 16 + n16] = f2bf(p);
        }
        __syncthreads();

        // ---- stage 2: acc(16x128) += P(16x128) @ W(128x128 slice of H)
        // B fragments come straight from transposed weights (contiguous in v)
#pragma unroll 1
        for (int k0 = 0; k0 < GCHUNK; k0 += 32) {
            Frag16 aP;
            const unsigned short* pr = Pl + n16 * GCHUNK + k0;
            aP.q[0] = *(const uint4*)(pr + ab1);
            aP.q[1] = *(const uint4*)(pr + ab2);
#pragma unroll
            for (int nt = 0; nt < 8; ++nt) {
                Frag16 bW;
                const int hcol = hbase_w + nt * 16 + n16;
                const unsigned short* tp = Wth + (size_t)hcol * VPAD + vbase + k0 + kbo;
                bW.q[0] = *(const uint4*)(tp);
                bW.q[1] = *(const uint4*)(tp + 8);
                acc[nt] = wmma_bf16(aP.v, bW.v, acc[nt]);   // 8 independent chains
            }
        }
    }

    // accumulate into global grad (other head + onehot terms add atomically)
    float* grad = out + 1;
#pragma unroll
    for (int nt = 0; nt < 8; ++nt) {
#pragma unroll
        for (int r = 0; r < 8; ++r) {
            const int m = r + hi8;
            const int hcol = hbase_w + nt * 16 + n16;
            atomicAdd(&grad[((size_t)(rb * 16 + m)) * DH + hcol], acc[nt][r]);
        }
    }
}

// ---------------- host-side launch ----------------
extern "C" void kernel_launch(void* const* d_in, const int* in_sizes, int n_in,
                              void* d_out, int out_size, void* d_ws, size_t ws_size,
                              hipStream_t stream) {
    const float* X      = (const float*)d_in[0];   // (B,T,H) f32
    const int*   labels = (const int*)  d_in[1];   // (B,T+2)
    const float* W      = (const float*)d_in[2];   // (2,V,H) f32
    const float* biases = (const float*)d_in[3];   // (2,V)   f32
    float* out = (float*)d_out;                    // [loss, grad(B,T,H)]

    // workspace layout (~424 MB)
    unsigned short* Xb  = (unsigned short*)d_ws;                // 4096*1024 bf16
    unsigned short* Wb  = Xb + (size_t)NROWS * DH;              // 2*V*1024 bf16
    unsigned short* Wbt = Wb + (size_t)NHEADS * DV * DH;        // 2*1024*VPAD bf16
    float* pm  = (float*)(Wbt + (size_t)NHEADS * DH * VPAD);    // 2*50*4096 f32
    float* ps  = pm + (size_t)NHEADS * NSCHUNK * NROWS;
    float* lse = ps + (size_t)NHEADS * NSCHUNK * NROWS;         // 2*4096 f32
    (void)in_sizes; (void)n_in; (void)ws_size;

    const int nX = NROWS * DH;
    const int nW = NHEADS * DV * DH;
    const int nOut = out_size;   // 1 + NROWS*DH

    zero_f32_kernel<<<dim3((nOut + 255) / 256), dim3(256), 0, stream>>>(out, nOut);
    cvt_bf16_kernel<<<dim3((nX + 255) / 256), dim3(256), 0, stream>>>(X, Xb, nX);
    cvt_bf16_kernel<<<dim3((nW + 255) / 256), dim3(256), 0, stream>>>(W, Wb, nW);
    transpose_w_kernel<<<dim3(VPAD / 32, DH / 32, NHEADS), dim3(32, 8), 0, stream>>>(W, Wbt);

    (void)hipFuncSetAttribute((const void*)stats_kernel,
                              hipFuncAttributeMaxDynamicSharedMemorySize, (int)STATS_LDS);
    stats_kernel<<<dim3(NROWS / 16, NSCHUNK, NHEADS), dim3(256), STATS_LDS, stream>>>(
        Xb, Wb, biases, pm, ps);

    lse_kernel<<<dim3((NHEADS * NROWS + 255) / 256), dim3(256), 0, stream>>>(pm, ps, lse);

    label_kernel<<<dim3(NROWS, NHEADS), dim3(256), 0, stream>>>(
        X, labels, W, biases, lse, out);

    (void)hipFuncSetAttribute((const void*)grad_kernel,
                              hipFuncAttributeMaxDynamicSharedMemorySize, (int)GRAD_LDS);
    grad_kernel<<<dim3(NROWS / 16, NHEADS), dim3(256), GRAD_LDS, stream>>>(
        Xb, Wb, Wbt, biases, lse, out);
}
